// MyPHlayer_80504866996719
// MI455X (gfx1250) — compile-verified, matrix operand order checked by reading, not run on hardware
//
#include <hip/hip_runtime.h>

// ---------------------------------------------------------------------------
// MI455X / gfx1250 implementation of the persistent-homology layer.
// wave32; WMMA 16x16x32 f16->f32 for all dense GEMMs with operands stored
// pre-swizzled in WMMA lane order (2x global_load_b128 per operand);
// Tensor Data Mover stages the D block into LDS for the triangle closure.
// ---------------------------------------------------------------------------

typedef _Float16 half_t;
typedef __attribute__((ext_vector_type(16))) _Float16     v16h;
typedef __attribute__((ext_vector_type(8)))  float        v8f;
typedef __attribute__((ext_vector_type(4)))  unsigned int v4u;
typedef __attribute__((ext_vector_type(8)))  int          v8i;
typedef __attribute__((ext_vector_type(4)))  int          v4i;

#define NROI   100
#define NB     4
#define BATCH  8
#define RESN   1000
#define RESP   1024      // padded grid rows (multiple of 32 for WMMA K)
#define DEGP   512       // padded coef rows (501 -> 512)
#define NITER  100       // Landweber iterations for the LS fit
#define BIGF   1e9f
#define SQRT2F 1.41421356237309515f

static __device__ __forceinline__ float ipow(float x, int k) {
  float r = 1.0f, p = x;
  while (k) { if (k & 1) r *= p; p *= p; k >>= 1; }
  return r;
}

// --- WMMA lane-order swizzles (ISA 7.12.2, wave32) --------------------------
// f16 A 16x32: lane L: m=L%16, hh=L/16; VGPR i<4 -> K=2i+8*hh(+j), i>=4 -> 16+2(i-4)+8*hh(+j)
// ko bit decode: hh=(ko>>3)&1, i=((ko>>4)<<2)|((ko>>1)&3), j=ko&1
static __device__ __forceinline__ size_t swA(int row, int col, int KC) {
  const int tile = row >> 4, m = row & 15;
  const int kc = col >> 5, ko = col & 31;
  const int hh = (ko >> 3) & 1;
  const int i  = ((ko >> 4) << 2) | ((ko >> 1) & 3);
  const int j  = ko & 1;
  const int lane = (hh << 4) | m;
  return ((((size_t)tile * KC + kc) * 32 + lane) << 4) + 2 * i + j;
}
// f16 B 32x16 mirrors A with n = column in lane 0..15
static __device__ __forceinline__ size_t swB(int row, int col) {
  const int kc = row >> 5, ko = row & 31;
  const int hh = (ko >> 3) & 1;
  const int i  = ((ko >> 4) << 2) | ((ko >> 1) & 3);
  const int j  = ko & 1;
  const int lane = (hh << 4) | col;
  return (((size_t)kc * 32 + lane) << 4) + 2 * i + j;
}
// swizzled operand loads: one aligned 32B block per lane -> 2x b128
static __device__ __forceinline__ v16h load_a_sw(const half_t* Asw, int KC, int tile, int kc) {
  const int lane = threadIdx.x & 31;
  return *(const v16h*)(Asw + ((((size_t)tile * KC + kc) * 32 + lane) << 4));
}
static __device__ __forceinline__ v16h load_b_sw(const half_t* Bsw, int kc) {
  const int lane = threadIdx.x & 31;
  return *(const v16h*)(Bsw + (((size_t)kc * 32 + lane) << 4));
}
// C/D: v8f, lane L col n=L%16, VGPR r -> row r + 8*(L/16)

// ---------------------------------------------------------------------------
__global__ void init_kernel(float* Dbuf, float* coefF, half_t* coefH,
                            int* cnt, unsigned* sminb, unsigned* smaxb) {
  const int gid = blockIdx.x * blockDim.x + threadIdx.x;
  const int str = gridDim.x * blockDim.x;
  for (int i = gid; i < NB * NROI * NROI; i += str) Dbuf[i] = 0.0f;
  for (int i = gid; i < DEGP * 16; i += str) { coefF[i] = 0.0f; coefH[i] = (half_t)0.0f; }
  if (gid < NB) { cnt[gid] = 0; sminb[gid] = 0x7F800000u; smaxb[gid] = 0u; }
}

__global__ void vand_kernel(half_t* VandH, half_t* VandTH, half_t* VdH) {
  const int gid = blockIdx.x * blockDim.x + threadIdx.x;
  if (gid >= RESP * DEGP) return;
  const int r = gid >> 9, k = gid & 511;
  float vx = 0.0f, dv = 0.0f;
  if (r < RESN) {
    const float x = 2.0f * (float)r / 999.0f - 1.0f;
    if (k <= 500) vx = ipow(x, k);
    if (k >= 1 && k <= 500) dv = (float)k * (2.0f / 999.0f) * ipow(x, k - 1);
  }
  VandH[swA(r, k, DEGP / 32)]  = (half_t)vx;   // V   : M=1024, K=512
  VdH[swA(r, k, DEGP / 32)]    = (half_t)dv;   // V'  : M=1024, K=512
  VandTH[swA(k, r, RESP / 32)] = (half_t)vx;   // V^T : M=512,  K=1024
}

__global__ void scatter_kernel(const int* ei, const float* ea, float* Dbuf, int E) {
  const int e = blockIdx.x * blockDim.x + threadIdx.x;
  if (e >= E) return;
  const int u = ei[e], v = ei[E + e];
  const int bu = u / NROI, bv = v / NROI;
  if (bu == bv)
    atomicAdd(&Dbuf[bu * NROI * NROI + (u % NROI) * NROI + (v % NROI)], ea[e]);
}

__global__ void dist_kernel(float* Dbuf) {
  const int i = blockIdx.x * blockDim.x + threadIdx.x;
  if (i < NB * NROI * NROI) Dbuf[i] = fmaxf(1.0f - Dbuf[i], 0.0f);
}

// --- Prim MST, one workgroup per block, serial 99 steps with LDS argmin -----
__global__ __launch_bounds__(128) void prim_kernel(const float* Dbuf, float* deaths,
                                                   float* maxd, unsigned char* mst) {
  __shared__ float mind[128];
  __shared__ int   par[128];
  __shared__ int   intree[128];
  __shared__ float rv[128];
  __shared__ int   ri[128];
  __shared__ float deathsS[128];
  __shared__ int   jstar_s;
  const int b = blockIdx.x, tid = threadIdx.x;
  const float* D = Dbuf + b * NROI * NROI;
  for (int p = tid; p < NROI * NROI; p += 128) mst[b * NROI * NROI + p] = 0;
  mind[tid] = (tid < NROI) ? D[tid] : BIGF;
  par[tid] = 0;
  intree[tid] = (tid == 0) ? 1 : 0;
  __syncthreads();
  for (int step = 0; step < NROI - 1; ++step) {
    rv[tid] = (tid < NROI && !intree[tid]) ? mind[tid] : BIGF;
    ri[tid] = tid;
    __syncthreads();
    for (int s = 64; s > 0; s >>= 1) {
      if (tid < s) {
        float v2 = rv[tid + s]; int i2 = ri[tid + s];
        if (v2 < rv[tid] || (v2 == rv[tid] && i2 < ri[tid])) { rv[tid] = v2; ri[tid] = i2; }
      }
      __syncthreads();
    }
    if (tid == 0) {
      const int js = ri[0]; jstar_s = js;
      const int u = par[js];
      deaths[b * 128 + step] = rv[0];
      deathsS[step] = rv[0];
      intree[js] = 1;
      mst[b * NROI * NROI + u * NROI + js] = 1;
      mst[b * NROI * NROI + js * NROI + u] = 1;
    }
    __syncthreads();
    const int js = jstar_s;
    if (tid < NROI && !intree[tid]) {
      const float dj = D[js * NROI + tid];
      if (dj < mind[tid]) { mind[tid] = dj; par[tid] = js; }
    }
    __syncthreads();
  }
  rv[tid] = (tid < NROI - 1) ? deathsS[tid] : -BIGF;
  __syncthreads();
  for (int s = 64; s > 0; s >>= 1) {
    if (tid < s) rv[tid] = fmaxf(rv[tid], rv[tid + s]);
    __syncthreads();
  }
  if (tid == 0) maxd[b] = rv[0];
}

__global__ void betti_kernel(const float* deaths, const float* maxd, float* Betti) {
  const int gid = blockIdx.x * blockDim.x + threadIdx.x;
  if (gid >= RESP * 16) return;
  const int t = gid >> 4, c = gid & 15;
  float v = 0.0f;
  if (t < RESN && c < NB) {
    const float g = maxd[c] * (float)t / 999.0f;
    int cv = 0;
#pragma unroll 1
    for (int i = 0; i < NROI - 1; ++i) cv += (g < deaths[c * 128 + i]) ? 1 : 0;
    v = (float)cv;
  }
  Betti[t * 16 + c] = v;
}

// --- Persistent LS-fit kernel: Landweber iterations, all GEMMs via WMMA -----
__global__ __launch_bounds__(256) void fit_kernel(const half_t* __restrict__ VandH,
                                                  const half_t* __restrict__ VandTH,
                                                  const float* __restrict__ Betti,
                                                  float* coefF, half_t* coefH,
                                                  half_t* resH) {
  const int w = threadIdx.x >> 5;
  const int lane = threadIdx.x & 31;
  const int n = lane & 15, hh = lane >> 4;
  const float omega = 1.0f / (501.0f * 1000.0f);  // 1/(||V||_1 * ||V||_inf)
  for (int it = 0; it < NITER; ++it) {
    // r = betti - V*c   (M = 1024, K = 512, N = 16)
    for (int tile = w; tile < RESP / 16; tile += 8) {
      v8f acc = {0.f, 0.f, 0.f, 0.f, 0.f, 0.f, 0.f, 0.f};
#pragma unroll 4
      for (int kc = 0; kc < DEGP / 32; ++kc) {
        __builtin_prefetch(VandH + ((((size_t)tile * (DEGP / 32) + kc + 1) * 32 + lane) << 4), 0, 1);
        v16h a  = load_a_sw(VandH, DEGP / 32, tile, kc);
        v16h bb = load_b_sw(coefH, kc);
        acc = __builtin_amdgcn_wmma_f32_16x16x32_f16(false, a, false, bb,
                                                     (short)0, acc, false, false);
      }
#pragma unroll
      for (int r = 0; r < 8; ++r) {
        const int row = tile * 16 + r + 8 * hh;
        resH[swB(row, n)] = (half_t)(Betti[row * 16 + n] - acc[r]);
      }
    }
    __threadfence_block();
    __syncthreads();
    // c += omega * V^T * r   (M = 512, K = 1024, N = 16)
    for (int tile = w; tile < DEGP / 16; tile += 8) {
      v8f acc = {0.f, 0.f, 0.f, 0.f, 0.f, 0.f, 0.f, 0.f};
#pragma unroll 4
      for (int kc = 0; kc < RESP / 32; ++kc) {
        __builtin_prefetch(VandTH + ((((size_t)tile * (RESP / 32) + kc + 1) * 32 + lane) << 4), 0, 1);
        v16h a  = load_a_sw(VandTH, RESP / 32, tile, kc);
        v16h bb = load_b_sw(resH, kc);
        acc = __builtin_amdgcn_wmma_f32_16x16x32_f16(false, a, false, bb,
                                                     (short)0, acc, false, false);
      }
#pragma unroll
      for (int r = 0; r < 8; ++r) {
        const int row = tile * 16 + r + 8 * hh;
        const float c2 = coefF[row * 16 + n] + omega * acc[r];
        coefF[row * 16 + n] = c2;
        coefH[swB(row, n)] = (half_t)c2;
      }
    }
    __threadfence_block();
    __syncthreads();
  }
}

// --- yvals = V*c ; fderiv = |V'*c|, written tiled x8 into d_out -------------
__global__ __launch_bounds__(32) void eval_kernel(const half_t* __restrict__ VandH,
                                                  const half_t* __restrict__ VdH,
                                                  const half_t* __restrict__ coefH,
                                                  float* out) {
  const int tile = blockIdx.x;
  const int which = blockIdx.y;  // 0 -> yvals, 1 -> fderiv
  const half_t* A = which ? VdH : VandH;
  v8f acc = {0.f, 0.f, 0.f, 0.f, 0.f, 0.f, 0.f, 0.f};
#pragma unroll 4
  for (int kc = 0; kc < DEGP / 32; ++kc) {
    v16h a  = load_a_sw(A, DEGP / 32, tile, kc);
    v16h bb = load_b_sw(coefH, kc);
    acc = __builtin_amdgcn_wmma_f32_16x16x32_f16(false, a, false, bb,
                                                 (short)0, acc, false, false);
  }
  const int lane = threadIdx.x & 31;
  const int n = lane & 15, hh = lane >> 4;
  const int YOFF = BATCH * NB * RESN;          // 32000
  const int FOFF = 2 * BATCH * NB * RESN;      // 64000
#pragma unroll
  for (int r = 0; r < 8; ++r) {
    const int row = tile * 16 + r + 8 * hh;
    if (n < NB) {
      if (which == 0) {
        if (row < RESN)
          for (int c = 0; c < BATCH; ++c) out[YOFF + (c * NB + n) * RESN + row] = acc[r];
      } else {
        if (row >= 50 && row < RESN - 50) {
          const float av = fabsf(acc[r]);
          for (int c = 0; c < BATCH; ++c) out[FOFF + (c * NB + n) * 900 + row - 50] = av;
        }
      }
    }
  }
}

// --- H1: tropical triangle closure T[i,j] = min_{k!=i,j} max(D_ik, D_jk) ----
// D block staged into LDS by the Tensor Data Mover (2D tile descriptor).
__global__ __launch_bounds__(256) void tri_kernel(const float* Dbuf, float* death1) {
  __shared__ float Ds[NROI * NROI];
  const int b = blockIdx.x;
  const float* src = Dbuf + b * NROI * NROI;
#if __has_builtin(__builtin_amdgcn_tensor_load_to_lds) && __has_builtin(__builtin_amdgcn_s_wait_tensorcnt)
  if (threadIdx.x < 32) {  // one wave issues the TDM op (EXEC ignored by TDM)
    const unsigned lds_off = (unsigned)(size_t)&Ds[0];   // addr[31:0] == LDS offset
    const unsigned long long ga = (unsigned long long)src;
    v4u g0;
    g0[0] = 1u;                                          // count=1, user descriptor
    g0[1] = lds_off;                                     // lds_addr (bytes)
    g0[2] = (unsigned)(ga & 0xffffffffu);                // global_addr[31:0]
    g0[3] = (unsigned)((ga >> 32) & 0x01ffffffu) | (2u << 30);  // addr[56:32] | type=2
    v8i g1;
    g1[0] = (int)(2u << 16);        // data_size = 4 bytes; no multicast/flags
    g1[1] = (int)(NROI << 16);      // tensor_dim0[15:0] in bits 63:48
    g1[2] = (int)(NROI << 16);      // tensor_dim1[15:0] in bits 95:80
    g1[3] = (int)(NROI << 16);      // tile_dim0 in bits 127:112
    g1[4] = NROI;                   // tile_dim1 = 100, tile_dim2 = 0
    g1[5] = NROI;                   // tensor_dim0_stride[31:0] = 100
    g1[6] = 0;
    g1[7] = 0;
    v4i gz = {0, 0, 0, 0};
#if __clang_major__ >= 23
    v8i gz8 = {0, 0, 0, 0, 0, 0, 0, 0};
    __builtin_amdgcn_tensor_load_to_lds(g0, g1, gz, gz, gz8, 0);
#else
    __builtin_amdgcn_tensor_load_to_lds(g0, g1, gz, gz, 0);
#endif
    __builtin_amdgcn_s_wait_tensorcnt((short)0);
  }
  __syncthreads();
#else
  for (int p = threadIdx.x; p < NROI * NROI; p += 256) Ds[p] = src[p];
  __syncthreads();
#endif
#if __has_builtin(__builtin_amdgcn_s_cluster_barrier)
  __builtin_amdgcn_s_cluster_barrier();  // NOP when not dispatched as a cluster (ISA 6.6)
#endif
  for (int p = threadIdx.x; p < NROI * NROI; p += 256) {
    const int i = p / NROI, j = p % NROI;
    const float* ri = Ds + i * NROI;
    const float* rj = Ds + j * NROI;
    float best = BIGF;
    for (int k = 0; k < NROI; ++k) {
      const float mx = fmaxf(ri[k], rj[k]);
      best = (k == i || k == j) ? best : fminf(best, mx);
    }
    death1[b * NROI * NROI + p] = fmaxf(Ds[p], best);
  }
}

__global__ __launch_bounds__(256) void compact_kernel(const float* Dbuf, const float* death1,
                                                      const unsigned char* mst, float2* edges,
                                                      int* cnt, unsigned* sminb, unsigned* smaxb) {
  const int b = blockIdx.x;
  for (int p = threadIdx.x; p < NROI * NROI; p += 256) {
    const int i = p / NROI, j = p % NROI;
    if (j > i && !mst[b * NROI * NROI + p]) {
      const float be = Dbuf[b * NROI * NROI + p];
      const float de = death1[b * NROI * NROI + p];
      const int idx = atomicAdd(&cnt[b], 1);
      edges[b * 5120 + idx] = make_float2(be, de);
      atomicMin(&sminb[b], __float_as_uint(be));   // values >= 0: uint order == float order
      atomicMax(&smaxb[b], __float_as_uint(de));
    }
  }
}

__global__ __launch_bounds__(256) void land_kernel(const float2* edges, const int* cnt,
                                                   const unsigned* sminb, const unsigned* smaxb,
                                                   float* out) {
  __shared__ float2 ch[1024];
  const int b = blockIdx.x;
  const int t = blockIdx.y * 256 + threadIdx.x;
  const bool act = t < RESN;
  const float smin = __uint_as_float(sminb[b]);
  const float smax = __uint_as_float(smaxb[b]);
  const float g = smin + (smax - smin) * (float)t / 999.0f;
  float t0 = 0.f, t1 = 0.f, t2 = 0.f, t3 = 0.f, t4 = 0.f;
  const int nc = cnt[b];
  for (int base = 0; base < nc; base += 1024) {
    const int m = (nc - base < 1024) ? (nc - base) : 1024;
    __syncthreads();
    for (int e = threadIdx.x; e < m; e += 256) ch[e] = edges[b * 5120 + base + e];
    __syncthreads();
    if (act) {
      for (int e = 0; e < m; ++e) {
        const float2 bd = ch[e];
        float v = fminf(g - bd.x, bd.y - g);
        v = (v > 0.f) ? v * SQRT2F : 0.f;
        if (v > t4) {
          if (v > t0)      { t4 = t3; t3 = t2; t2 = t1; t1 = t0; t0 = v; }
          else if (v > t1) { t4 = t3; t3 = t2; t2 = t1; t1 = v; }
          else if (v > t2) { t4 = t3; t3 = t2; t2 = v; }
          else if (v > t3) { t4 = t3; t3 = v; }
          else             { t4 = v; }
        }
      }
    }
  }
  if (act) {
    const float land = (t0 + t1 + t2 + t3 + t4) * 0.2f;
    for (int c = 0; c < BATCH; ++c) out[(c * NB + b) * RESN + t] = land;
  }
}

// ---------------------------------------------------------------------------
extern "C" void kernel_launch(void* const* d_in, const int* in_sizes, int n_in,
                              void* d_out, int out_size, void* d_ws, size_t ws_size,
                              hipStream_t stream) {
  const int* ei = (const int*)d_in[1];
  const float* ea = (const float*)d_in[2];
  float* out = (float*)d_out;
  const int E = in_sizes[1] / 2;  // 160000

  char* ws = (char*)d_ws;
  size_t off = 0;
  auto alloc = [&](size_t bytes) { size_t o = off; off = (off + bytes + 255) & ~(size_t)255; return o; };
  float*         Dbuf   = (float*)(ws + alloc(NB * NROI * NROI * 4));
  half_t*        VandH  = (half_t*)(ws + alloc((size_t)RESP * DEGP * 2));
  half_t*        VandTH = (half_t*)(ws + alloc((size_t)DEGP * RESP * 2));
  half_t*        VdH    = (half_t*)(ws + alloc((size_t)RESP * DEGP * 2));
  float*         Betti  = (float*)(ws + alloc((size_t)RESP * 16 * 4));
  float*         coefF  = (float*)(ws + alloc((size_t)DEGP * 16 * 4));
  half_t*        coefH  = (half_t*)(ws + alloc((size_t)DEGP * 16 * 2));
  half_t*        resH   = (half_t*)(ws + alloc((size_t)RESP * 16 * 2));
  float*         deaths = (float*)(ws + alloc(NB * 128 * 4));
  float*         maxd   = (float*)(ws + alloc(NB * 4));
  unsigned char* mst    = (unsigned char*)(ws + alloc(NB * NROI * NROI));
  float*         death1 = (float*)(ws + alloc(NB * NROI * NROI * 4));
  float2*        edges  = (float2*)(ws + alloc(NB * 5120 * 8));
  int*           cnt    = (int*)(ws + alloc(NB * 4));
  unsigned*      sminb  = (unsigned*)(ws + alloc(NB * 4));
  unsigned*      smaxb  = (unsigned*)(ws + alloc(NB * 4));
  (void)ws_size; (void)n_in; (void)out_size;

  init_kernel<<<64, 256, 0, stream>>>(Dbuf, coefF, coefH, cnt, sminb, smaxb);
  vand_kernel<<<(RESP * DEGP + 255) / 256, 256, 0, stream>>>(VandH, VandTH, VdH);
  scatter_kernel<<<(E + 255) / 256, 256, 0, stream>>>(ei, ea, Dbuf, E);
  dist_kernel<<<(NB * NROI * NROI + 255) / 256, 256, 0, stream>>>(Dbuf);
  prim_kernel<<<NB, 128, 0, stream>>>(Dbuf, deaths, maxd, mst);
  betti_kernel<<<(RESP * 16 + 255) / 256, 256, 0, stream>>>(deaths, maxd, Betti);
  fit_kernel<<<1, 256, 0, stream>>>(VandH, VandTH, Betti, coefF, coefH, resH);
  eval_kernel<<<dim3(RESP / 16, 2), 32, 0, stream>>>(VandH, VdH, coefH, out);
  tri_kernel<<<NB, 256, 0, stream>>>(Dbuf, death1);
  compact_kernel<<<NB, 256, 0, stream>>>(Dbuf, death1, mst, edges, cnt, sminb, smaxb);
  land_kernel<<<dim3(NB, 4), 256, 0, stream>>>(edges, cnt, sminb, smaxb, out);
}
